// MultiplexedMOE_3272765079848
// MI455X (gfx1250) — matrix operand order repaired and checked by reading, not compiled, not run on hardware
//
#include <hip/hip_runtime.h>
#include <hip/hip_bf16.h>

typedef __attribute__((ext_vector_type(16))) __bf16 v16bf;
typedef __attribute__((ext_vector_type(8)))  __bf16 bf16x8;
typedef __attribute__((ext_vector_type(8)))  float  v8f;

#define HDIM 2048
#define NTOK 2048
#define ISZ  1408
#define GQ   8
#define NEXP 64
#define KB   64   // K columns per weight LDS buffer (2 WMMA sub-steps)
#define LDSW 72   // LDS row stride in bf16 (144B: 16B-aligned, bank-skewed)

// ---------------- WMMA helpers ----------------

__device__ __forceinline__ v8f wmma_bf16(v16bf a, v16bf b, v8f c) {
  return __builtin_amdgcn_wmma_f32_16x16x32_bf16(
      /*neg_a=*/false, a, /*neg_b=*/false, b,
      /*c_mod=*/(short)0, c, /*reuse_a=*/false, /*reuse_b=*/false);
}

__device__ __forceinline__ v16bf make16(bf16x8 lo, bf16x8 hi) {
  v16bf r;
#pragma unroll
  for (int j = 0; j < 8; ++j) { r[j] = lo[j]; r[j + 8] = hi[j]; }
  return r;
}

// A-matrix 16x32 bf16 fragment loaded DIRECTLY from a global row-major matrix.
// rowPtr = &A[mBase + row16][0] (global), kbase = start of 32-wide K window.
__device__ __forceinline__ v16bf load_afrag_g(const __bf16* rowPtr, int kbase,
                                              int half) {
  bf16x8 lo = *(const bf16x8*)(rowPtr + kbase + half * 8);
  bf16x8 hi = *(const bf16x8*)(rowPtr + kbase + 16 + half * 8);
  return make16(lo, hi);
}

// B-matrix 32x16 bf16 fragment from LDS (LDS holds B^T: [n][k]).
__device__ __forceinline__ v16bf load_bfrag(const __bf16* base, int nBase,
                                            int row16, int half) {
  const __bf16* p = base + (size_t)(nBase + row16) * LDSW + half * 8;
  bf16x8 lo = *(const bf16x8*)(p);
  bf16x8 hi = *(const bf16x8*)(p + 16);
  return make16(lo, hi);
}

__device__ __forceinline__ bf16x8 cvt8(float4 a, float4 b) {
  bf16x8 r;
  r[0] = (__bf16)a.x; r[1] = (__bf16)a.y; r[2] = (__bf16)a.z; r[3] = (__bf16)a.w;
  r[4] = (__bf16)b.x; r[5] = (__bf16)b.y; r[6] = (__bf16)b.z; r[7] = (__bf16)b.w;
  return r;
}

// ---------------- Kernel 1: routing / gating ----------------
__global__ __launch_bounds__(64) void gating_kernel(
    const float* __restrict__ x, const float* __restrict__ gate_w,
    float* __restrict__ soft, float* __restrict__ scalar_out) {
  __shared__ float s_xr[HDIM];
  __shared__ float s_red[NEXP];
  __shared__ float s_prob[NEXP];
  const int n = blockIdx.x;
  const int t = threadIdx.x;

  for (int h = t; h < HDIM; h += 64) s_xr[h] = x[(size_t)n * HDIM + h];
  __syncthreads();

  const float* wrow = gate_w + (size_t)t * HDIM;
  float acc = 0.f;
  for (int h = 0; h < HDIM; ++h) acc = fmaf(s_xr[h], wrow[h], acc);
  s_red[t] = acc;
  __syncthreads();

  float mx = -3.0e38f;
  for (int e = 0; e < NEXP; ++e) mx = fmaxf(mx, s_red[e]);
  float p = __expf(acc - mx);
  s_prob[t] = p;
  __syncthreads();
  float sum = 0.f;
  for (int e = 0; e < NEXP; ++e) sum += s_prob[e];
  __syncthreads();
  s_prob[t] = p / sum;
  __syncthreads();

  if (t == 0) {
    float tmp[NEXP], flat[NEXP];
    for (int e = 0; e < NEXP; ++e) { tmp[e] = s_prob[e]; flat[e] = 0.f; }
    for (int k = 0; k < 8; ++k) {  // top-8
      int best = 0; float bv = tmp[0];
      for (int e = 1; e < NEXP; ++e)
        if (tmp[e] > bv) { bv = tmp[e]; best = e; }
      flat[best] = bv;
      tmp[best] = -1.f;
    }
    for (int g = 0; g < GQ; ++g) {
      float s = 0.f, m = -3.0e38f, v[8];
      for (int j = 0; j < 8; ++j) {
        float f = flat[g * 8 + j];
        s += f;
        v[j] = (f == 0.f) ? -1.0e9f : f;
        m = fmaxf(m, v[j]);
      }
      scalar_out[(size_t)n * GQ + g] = s;
      float den = 0.f, ex[8];
      for (int j = 0; j < 8; ++j) { ex[j] = __expf(v[j] - m); den += ex[j]; }
      float rden = __builtin_amdgcn_rcpf(den);
      for (int j = 0; j < 8; ++j)
        soft[(size_t)n * NEXP + g * 8 + j] = ex[j] * rden;
    }
  }
}

// ---------------- Kernel 2: xg = x + soft . mask_up^T (bf16, layout [g][n][h]) ----------------
__global__ __launch_bounds__(256) void xg_kernel(
    const float* __restrict__ x, const float* __restrict__ muw,
    const float* __restrict__ soft, __bf16* __restrict__ xg) {
  const int g = blockIdx.z;
  const int n = blockIdx.y;
  const int h = blockIdx.x * 256 + threadIdx.x;
  const float* sp = soft + (size_t)n * NEXP + g * 8;   // uniform -> scalar loads
  const float* mp = muw + ((size_t)g * HDIM + h) * 8;  // 8 contiguous f32
  float c = 0.f;
#pragma unroll
  for (int s = 0; s < 8; ++s) c = fmaf(sp[s], mp[s], c);
  xg[((size_t)g * NTOK + n) * HDIM + h] = (__bf16)(x[(size_t)n * HDIM + h] + c);
}

// ---------------- Kernel 3: gate/up GEMMs + SiLU + per-(token,group) scalar ----------------
// tile: 128 tokens x 64 i. A-fragments (xg, bf16) loaded straight from
// global/L2 (with one-step-ahead global_prefetch); weight tiles
// double-buffered in LDS (f32->bf16), 1 barrier/step.
// act[g][tok][i] = scalar[tok,g] * silu(gate) * up   (bf16)
__global__ __launch_bounds__(256, 1) void gateup_kernel(
    const __bf16* __restrict__ xg, const float* __restrict__ wg,
    const float* __restrict__ wu, const float* __restrict__ scal,
    __bf16* __restrict__ act) {
  __shared__ __bf16 s_g[2][64 * LDSW];
  __shared__ __bf16 s_u[2][64 * LDSW];
  __shared__ float s_scal[128];
  const int t = threadIdx.x;
  const int g = blockIdx.z;
  const int n0 = blockIdx.y * 128;
  const int i0 = blockIdx.x * 64;
  const int lane = t & 31, w = t >> 5;
  const int row16 = lane & 15, half = lane >> 4;
  const int wm = (w & 3) * 32;   // token stripe
  const int wn = (w >> 2) * 32;  // i stripe

  v8f accG[2][2] = {{{}, {}}, {{}, {}}};
  v8f accU[2][2] = {{{}, {}}, {{}, {}}};

  // per-lane global A row pointers (mi=0 / mi=1 sub-tiles)
  const __bf16* aRow0 = xg + ((size_t)(g * NTOK + n0 + wm + row16)) * HDIM;
  const __bf16* aRow1 = aRow0 + (size_t)16 * HDIM;

  // weight staging geometry: 64x64 f32 tile, 16 f32 per thread
  const int wr = t >> 2, wseg = (t & 3) * 16;
  const float* wgBase = wg + ((size_t)(g * ISZ + i0 + wr)) * HDIM + wseg;
  const float* wuBase = wu + ((size_t)(g * ISZ + i0 + wr)) * HDIM + wseg;

  float4 pg[4], pu[4];
  auto prefetch = [&](int k0) {
    const float4* g4 = (const float4*)(wgBase + k0);
    pg[0] = g4[0]; pg[1] = g4[1]; pg[2] = g4[2]; pg[3] = g4[3];
    const float4* u4 = (const float4*)(wuBase + k0);
    pu[0] = u4[0]; pu[1] = u4[1]; pu[2] = u4[2]; pu[3] = u4[3];
  };
  auto commit = [&](int buf) {
    __bf16* gd = &s_g[buf][wr * LDSW + wseg];
    *(bf16x8*)(gd) = cvt8(pg[0], pg[1]);
    *(bf16x8*)(gd + 8) = cvt8(pg[2], pg[3]);
    __bf16* ud = &s_u[buf][wr * LDSW + wseg];
    *(bf16x8*)(ud) = cvt8(pu[0], pu[1]);
    *(bf16x8*)(ud + 8) = cvt8(pu[2], pu[3]);
  };

  if (t < 128) s_scal[t] = scal[(size_t)(n0 + t) * GQ + g];
  prefetch(0);
  commit(0);
  __syncthreads();

  const int steps = HDIM / KB;  // 32
  for (int s = 0; s < steps; ++s) {
    const int buf = s & 1;
    if (s + 1 < steps) {
      prefetch((s + 1) * KB);
      // warm L0/L2 for next step's direct-from-global A fragments
      __builtin_prefetch(aRow0 + (s + 1) * KB, 0, 3);
      __builtin_prefetch(aRow1 + (s + 1) * KB, 0, 3);
    }
#pragma unroll
    for (int kk = 0; kk < KB; kk += 32) {
      const int kbase = s * KB + kk;
      {
        v16bf aF = load_afrag_g(aRow0, kbase, half);
#pragma unroll
        for (int ni = 0; ni < 2; ++ni) {
          v16bf bG = load_bfrag(&s_g[buf][kk], wn + ni * 16, row16, half);
          accG[0][ni] = wmma_bf16(aF, bG, accG[0][ni]);
          v16bf bU = load_bfrag(&s_u[buf][kk], wn + ni * 16, row16, half);
          accU[0][ni] = wmma_bf16(aF, bU, accU[0][ni]);
        }
      }
      {
        v16bf aF = load_afrag_g(aRow1, kbase, half);
#pragma unroll
        for (int ni = 0; ni < 2; ++ni) {
          v16bf bG = load_bfrag(&s_g[buf][kk], wn + ni * 16, row16, half);
          accG[1][ni] = wmma_bf16(aF, bG, accG[1][ni]);
          v16bf bU = load_bfrag(&s_u[buf][kk], wn + ni * 16, row16, half);
          accU[1][ni] = wmma_bf16(aF, bU, accU[1][ni]);
        }
      }
    }
    if (s + 1 < steps) {
      commit(buf ^ 1);
      __syncthreads();
    }
  }

  // epilogue: act = scalar * silu(gate) * up -> bf16 act[g][tok][i]
#pragma unroll
  for (int mi = 0; mi < 2; ++mi)
#pragma unroll
    for (int ni = 0; ni < 2; ++ni) {
      const int iCol = i0 + wn + ni * 16 + row16;
      v8f gA = accG[mi][ni], uA = accU[mi][ni];
#pragma unroll
      for (int v = 0; v < 8; ++v) {
        int trow = wm + mi * 16 + half * 8 + v;
        float gv = gA[v], uv = uA[v];
        float av = gv * __builtin_amdgcn_rcpf(1.f + __expf(-gv)) * uv;
        av *= s_scal[trow];
        act[((size_t)(g * NTOK + n0 + trow)) * ISZ + iCol] = (__bf16)av;
      }
    }
}

// ---------------- Kernel 4: y[n][h] = sum_g (act_g . Wd_g^T) ----------------
// (scalar already folded into act). A-fragments from global/L2; Wd tiles
// double-buffered in LDS; flattened (g,k) loop; f32 store fully overwrites y.
__global__ __launch_bounds__(256, 1) void down_kernel(
    const __bf16* __restrict__ act, const float* __restrict__ wd,
    float* __restrict__ y) {
  __shared__ __bf16 s_d[2][64 * LDSW];
  const int t = threadIdx.x;
  const int hb = blockIdx.x * 64;
  const int n0 = blockIdx.y * 128;
  const int lane = t & 31, w = t >> 5;
  const int row16 = lane & 15, half = lane >> 4;
  const int wm = (w & 3) * 32;
  const int wn = (w >> 2) * 32;

  v8f acc[2][2] = {{{}, {}}, {{}, {}}};

  const int wr = t >> 2, wseg = (t & 3) * 16;

  const int kstepsPerG = ISZ / KB;    // 22
  const int steps = GQ * kstepsPerG;  // 176

  float4 pd[4];
  auto prefetch = [&](int s) {
    const int gg = s / kstepsPerG;
    const int k0 = (s % kstepsPerG) * KB;
    const float4* d4 =
        (const float4*)(wd + ((size_t)(gg * HDIM + hb + wr)) * ISZ + wseg + k0);
    pd[0] = d4[0]; pd[1] = d4[1]; pd[2] = d4[2]; pd[3] = d4[3];
  };
  auto commit = [&](int buf) {
    __bf16* dd = &s_d[buf][wr * LDSW + wseg];
    *(bf16x8*)(dd) = cvt8(pd[0], pd[1]);
    *(bf16x8*)(dd + 8) = cvt8(pd[2], pd[3]);
  };

  prefetch(0);
  commit(0);
  __syncthreads();

  for (int s = 0; s < steps; ++s) {
    const int buf = s & 1;
    const int gg = s / kstepsPerG;
    const int k0 = (s % kstepsPerG) * KB;
    const __bf16* aRow0 =
        act + ((size_t)(gg * NTOK + n0 + wm + row16)) * ISZ + k0;
    const __bf16* aRow1 = aRow0 + (size_t)16 * ISZ;
    if (s + 1 < steps) {
      prefetch(s + 1);
      const int gn = (s + 1) / kstepsPerG;
      const int kn = ((s + 1) % kstepsPerG) * KB;
      const __bf16* nRow =
          act + ((size_t)(gn * NTOK + n0 + wm + row16)) * ISZ + kn;
      __builtin_prefetch(nRow, 0, 3);
      __builtin_prefetch(nRow + (size_t)16 * ISZ, 0, 3);
    }
#pragma unroll
    for (int kk = 0; kk < KB; kk += 32) {
      {
        v16bf aF = load_afrag_g(aRow0, kk, half);
#pragma unroll
        for (int ni = 0; ni < 2; ++ni) {
          v16bf bD = load_bfrag(&s_d[buf][kk], wn + ni * 16, row16, half);
          acc[0][ni] = wmma_bf16(aF, bD, acc[0][ni]);
        }
      }
      {
        v16bf aF = load_afrag_g(aRow1, kk, half);
#pragma unroll
        for (int ni = 0; ni < 2; ++ni) {
          v16bf bD = load_bfrag(&s_d[buf][kk], wn + ni * 16, row16, half);
          acc[1][ni] = wmma_bf16(aF, bD, acc[1][ni]);
        }
      }
    }
    if (s + 1 < steps) {
      commit(buf ^ 1);
      __syncthreads();
    }
  }

#pragma unroll
  for (int mi = 0; mi < 2; ++mi)
#pragma unroll
    for (int ni = 0; ni < 2; ++ni) {
      const int hCol = hb + wn + ni * 16 + row16;
#pragma unroll
      for (int v = 0; v < 8; ++v) {
        int tok = n0 + wm + mi * 16 + half * 8 + v;
        y[(size_t)tok * HDIM + hCol] = acc[mi][ni][v];
      }
    }
}

// ---------------- launch ----------------
extern "C" void kernel_launch(void* const* d_in, const int* in_sizes, int n_in,
                              void* d_out, int out_size, void* d_ws, size_t ws_size,
                              hipStream_t stream) {
  const float* x           = (const float*)d_in[0];
  const float* gate_w      = (const float*)d_in[1];
  const float* mask_up_w   = (const float*)d_in[2];
  const float* gate_proj_w = (const float*)d_in[3];
  const float* up_proj_w   = (const float*)d_in[4];
  const float* down_proj_w = (const float*)d_in[5];
  float* y = (float*)d_out;

  char* ws = (char*)d_ws;
  float* soft = (float*)ws;                           // N*64 f32   = 512 KB
  float* scal = soft + (size_t)NTOK * NEXP;           // N*8  f32   = 64 KB
  __bf16* xg  = (__bf16*)(scal + (size_t)NTOK * GQ);  // G*N*H bf16 = 64 MB
  __bf16* act = xg + (size_t)GQ * NTOK * HDIM;        // G*N*I bf16 = 44 MB

  gating_kernel<<<NTOK, 64, 0, stream>>>(x, gate_w, soft, scal);
  xg_kernel<<<dim3(HDIM / 256, NTOK, GQ), 256, 0, stream>>>(x, mask_up_w, soft, xg);
  gateup_kernel<<<dim3(ISZ / 64, NTOK / 128, GQ), 256, 0, stream>>>(
      xg, gate_proj_w, up_proj_w, scal, act);
  down_kernel<<<dim3(HDIM / 64, NTOK / 128), 256, 0, stream>>>(act, down_proj_w, y);
}
